// DiscreteContinuousConvS2_39316130627585
// MI455X (gfx1250) — matrix operand order repaired and checked by reading, not compiled, not run on hardware
//
#include <hip/hip_runtime.h>
#include <hip/hip_bf16.h>
#include <stdint.h>

// Problem constants (match reference)
#define NLAT  181
#define NLON  360
#define BATCH 4
#define CIN   96
#define COUT  96
#define KSIZE 2
#define NROWS (KSIZE * NLAT)       // 362 sparse output rows (k, ho)
#define S_TOT (NLAT * NLON)        // 65160 spatial points per batch
#define KTOT  (KSIZE * CIN)        // 192 contraction length

typedef __attribute__((ext_vector_type(16))) _Float16 v16h;
typedef __attribute__((ext_vector_type(8)))  _Float16 v8h;
typedef __attribute__((ext_vector_type(8)))  float    v8f;

// ---------------------------------------------------------------------------
// CSR construction. COO entries are emitted by the reference sorted by
// (ho, k), so each row g = ho*KSIZE + k is ONE contiguous range: we only need
// per-row counts + an exclusive scan. Fully deterministic (integer atomics
// only; entry order within a row is the original order).
// ---------------------------------------------------------------------------
__global__ void k_zero_counts(int* __restrict__ counts) {
    int i = blockIdx.x * blockDim.x + threadIdx.x;
    if (i < NROWS) counts[i] = 0;
}

__global__ void k_hist(const int* __restrict__ ho_idx, const int* __restrict__ k_idx,
                       int* __restrict__ counts, int nnz) {
    int j = blockIdx.x * blockDim.x + threadIdx.x;
    if (j < nnz) atomicAdd(&counts[ho_idx[j] * KSIZE + k_idx[j]], 1);
}

__global__ void k_scan(const int* __restrict__ counts, int* __restrict__ roff) {
    if (blockIdx.x == 0 && threadIdx.x == 0) {
        int a = 0;
        for (int i = 0; i < NROWS; ++i) { roff[i] = a; a += counts[i]; }
        roff[NROWS] = a;
    }
}

// ---------------------------------------------------------------------------
// Stage 1: sparse gather-accumulate.
//   y[b][s = ho*360+pout][kidx = k*96+c] = sum_j psi[j]*x[b,c,la_j,(lo_j+pout)%360]
// Block = (row g, batch b, 16-channel slab). 384 threads over pout:
//  - x loads: 32 lanes read consecutive lon -> fully coalesced 128B bursts,
//    and x (100MB) is L2-resident on MI455X (192MB L2), so the j-loop re-reads
//    are L2 hits, not HBM traffic.
//  - (val, la, lo) are wave-uniform -> scalar loads, amortized over 16
//    register accumulators per thread.
//  - LDS transpose turns the channel-strided f16 stores into 32B contiguous
//    bursts into the GEMM-friendly y[b][s][kidx] layout.
// ---------------------------------------------------------------------------
__global__ __launch_bounds__(384) void k_gather(
    const float* __restrict__ x, const float* __restrict__ psi,
    const int* __restrict__ lat_in, const int* __restrict__ lon_in,
    const int* __restrict__ roff, _Float16* __restrict__ y)
{
    __shared__ float sm[16][NLON + 1];   // +1 pad: conflict-free transpose

    const int g    = blockIdx.x;         // g = ho*KSIZE + k
    const int ho   = g / KSIZE;
    const int k    = g % KSIZE;
    const int b    = blockIdx.y;
    const int cblk = blockIdx.z;         // 6 slabs of 16 channels
    const int js   = roff[g];
    const int je   = roff[g + 1];
    const int p    = threadIdx.x;        // pout

    float acc[16];
#pragma unroll
    for (int i = 0; i < 16; ++i) acc[i] = 0.0f;

    if (p < NLON) {
        const float* xb = x + (size_t)(b * CIN + cblk * 16) * (NLAT * NLON);
        for (int j = js; j < je; ++j) {
            const float v  = psi[j];      // wave-uniform
            const int   la = lat_in[j];
            int lon = lon_in[j] + p;
            if (lon >= NLON) lon -= NLON;
            const float* xr = xb + (size_t)la * NLON + lon;
#pragma unroll
            for (int cc = 0; cc < 16; ++cc)
                acc[cc] += v * xr[(size_t)cc * (NLAT * NLON)];
        }
#pragma unroll
        for (int cc = 0; cc < 16; ++cc) sm[cc][p] = acc[cc];
    }
    __syncthreads();

    // Transposed store: 16 consecutive f16 per pout (32B contiguous).
    const int cc = threadIdx.x & 15;
    const int pr = threadIdx.x >> 4;     // 0..23
    const int kO = k * CIN + cblk * 16 + cc;
    for (int base = 0; base < NLON; base += 24) {
        const int pp = base + pr;        // covers 0..359 exactly
        y[((size_t)b * S_TOT + (size_t)ho * NLON + pp) * KTOT + kO] =
            (_Float16)sm[cc][pp];
    }
}

// ---------------------------------------------------------------------------
// Stage 2: GEMM with CDNA5 WMMA.
//   out[b][o][s] = sum_{kidx<192} Wt[o][kidx] * y[b][s][kidx] + bias[o]
// 8 waves/block; each wave owns a 96(M) x 32(N) tile: 12 v8f accumulators
// (two 16-wide N sub-tiles sharing every A fragment -> half the LDS traffic
// per WMMA and two independent WMMAs per ds_load pair, so the matrix pipe
// is not serialized on s_wait_dscnt). 6 k-steps x 6 m-tiles x 2 n-tiles =
// 72 v_wmma_f32_16x16x32_f16 per wave.
// Fragment packing uses contiguous k order (k = 16*half + i), same in A and B:
//   A (16x32): af[i]   = A[m][half*16 + i]  -> 16 contiguous f16 per lane
//   B (32x16): bf[2r+e]= B[half*16 + 2r+e][n]  -> 16 contiguous f16 per lane
// ---------------------------------------------------------------------------
__global__ __launch_bounds__(256) void k_gemm(
    const _Float16* __restrict__ y, const float* __restrict__ w,
    const float* __restrict__ bias, float* __restrict__ out)
{
    __shared__ _Float16 Wt[COUT][KTOT] __attribute__((aligned(16))); // 36 KB
    __shared__ float    bsh[COUT];

    const int tid = threadIdx.x;
    for (int t = tid; t < COUT * KTOT; t += 256) {
        const int o = t / KTOT, kidx = t % KTOT;
        const int k = kidx / CIN, c = kidx % CIN;
        Wt[o][kidx] = (_Float16)w[(o * CIN + c) * KSIZE + k];
    }
    if (tid < COUT) bsh[tid] = bias[tid];
    __syncthreads();

    const int wave = tid >> 5, lane = tid & 31;
    const int half = lane >> 4, n = lane & 15, m = lane & 15;
    const int b  = blockIdx.y;
    const int s0 = blockIdx.x * 256 + wave * 32;

    const int s_a = s0 + n;                              // N sub-tile 0
    const int s_b = s0 + 16 + n;                         // N sub-tile 1
    const int sca = (s_a < S_TOT) ? s_a : (S_TOT - 1);   // clamp loads
    const int scb = (s_b < S_TOT) ? s_b : (S_TOT - 1);
    const _Float16* yrow_a = y + ((size_t)b * S_TOT + sca) * KTOT;
    const _Float16* yrow_b = y + ((size_t)b * S_TOT + scb) * KTOT;

    v8f acc[12];
#pragma unroll
    for (int i = 0; i < 12; ++i)
#pragma unroll
        for (int r = 0; r < 8; ++r) acc[i][r] = 0.0f;

#pragma unroll
    for (int t = 0; t < 6; ++t) {
        const int kk = t * 32;

        // B fragments: 16 contiguous f16 = two 16B global loads per lane each.
        v16h bf0, bf1;
        {
            const v8h* bp = (const v8h*)(yrow_a + kk + half * 16);
            const v8h lo = bp[0], hi = bp[1];
#pragma unroll
            for (int i = 0; i < 8; ++i) { bf0[i] = lo[i]; bf0[8 + i] = hi[i]; }
        }
        {
            const v8h* bp = (const v8h*)(yrow_b + kk + half * 16);
            const v8h lo = bp[0], hi = bp[1];
#pragma unroll
            for (int i = 0; i < 8; ++i) { bf1[i] = lo[i]; bf1[8 + i] = hi[i]; }
        }
        if (t < 5) {  // pull next k-step's B data (global_prefetch_b8)
            __builtin_prefetch(yrow_a + kk + 32 + half * 16, 0, 1);
            __builtin_prefetch(yrow_b + kk + 32 + half * 16, 0, 1);
        }

#pragma unroll
        for (int mt = 0; mt < 6; ++mt) {
            const int rowm = mt * 16 + m;
            const v8h alo = *(const v8h*)&Wt[rowm][kk + half * 16];
            const v8h ahi = *(const v8h*)&Wt[rowm][kk + half * 16 + 8];
            v16h af;
#pragma unroll
            for (int i = 0; i < 8; ++i) { af[i] = alo[i]; af[8 + i] = ahi[i]; }

            // Two independent WMMAs share this A fragment.
            acc[2 * mt + 0] = __builtin_amdgcn_wmma_f32_16x16x32_f16(
                false, af, false, bf0, (short)0, acc[2 * mt + 0], false, false);
            acc[2 * mt + 1] = __builtin_amdgcn_wmma_f32_16x16x32_f16(
                false, af, false, bf1, (short)0, acc[2 * mt + 1], false, false);
        }
    }

#pragma unroll
    for (int u = 0; u < 2; ++u) {
        const int s = (u == 0) ? s_a : s_b;
        if (s < S_TOT) {
#pragma unroll
            for (int mt = 0; mt < 6; ++mt)
#pragma unroll
                for (int r = 0; r < 8; ++r) {
                    const int o = mt * 16 + r + 8 * half;  // D layout: M=r+8*half
                    out[((size_t)b * COUT + o) * S_TOT + s] =
                        acc[2 * mt + u][r] + bsh[o];
                }
        }
    }
}

// ---------------------------------------------------------------------------
extern "C" void kernel_launch(void* const* d_in, const int* in_sizes, int n_in,
                              void* d_out, int out_size, void* d_ws, size_t ws_size,
                              hipStream_t stream) {
    const float* x      = (const float*)d_in[0];
    const float* wgt    = (const float*)d_in[1];
    const float* bias   = (const float*)d_in[2];
    const float* psi    = (const float*)d_in[3];
    const int*   k_idx  = (const int*)d_in[4];
    const int*   ho_idx = (const int*)d_in[5];
    const int*   lat_in = (const int*)d_in[6];
    const int*   lon_in = (const int*)d_in[7];
    float*       out    = (float*)d_out;
    const int    nnz    = in_sizes[3];

    // Workspace: y (f16, 100MB, 16B-aligned at ws start) + CSR arrays.
    char* ws = (char*)d_ws;
    _Float16* y = (_Float16*)ws;
    const size_t ybytes = (size_t)BATCH * S_TOT * KTOT * sizeof(_Float16);
    const size_t off0   = (ybytes + 255) & ~(size_t)255;
    int* counts = (int*)(ws + off0);
    int* roff   = counts + NROWS;            // NROWS+1 ints

    k_zero_counts<<<(NROWS + 255) / 256, 256, 0, stream>>>(counts);
    k_hist<<<(nnz + 255) / 256, 256, 0, stream>>>(ho_idx, k_idx, counts, nnz);
    k_scan<<<1, 32, 0, stream>>>(counts, roff);
    k_gather<<<dim3(NROWS, BATCH, CIN / 16), 384, 0, stream>>>(
        x, psi, lat_in, lon_in, roff, y);
    k_gemm<<<dim3((S_TOT + 255) / 256, BATCH), 256, 0, stream>>>(
        y, wgt, bias, out);
}
